// MyModel_61933428415509
// MI455X (gfx1250) — matrix operand order repaired and checked by reading, not compiled, not run on hardware
//
#include <hip/hip_runtime.h>
#include <hip/hip_bf16.h>
#include <stdint.h>

// Problem dims
#define T_STEPS 1024
#define BATCH   128
#define IN_DIM  140
#define HID     340
// Packed dims
#define KDIM        512   // 352 (state, padded from 340) + 160 (input, padded from 140)
#define K_INPUT_OFF 352
#define NPAD        704   // two 352-wide halves (inp | gate), each padded from 340
#define NTILES_ALL  44    // 704/16
#define NTILES_HALF 22
#define KTILES      16    // 512/32
#define NWG         22    // one workgroup per column-tile pair
#define WAVES_PER_WG 8    // one wave per M-tile

typedef __attribute__((ext_vector_type(16))) __bf16 v16bf;
typedef __attribute__((ext_vector_type(8)))  float  v8f;
typedef __attribute__((ext_vector_type(4)))  unsigned int u32x4;

// Workspace layout (bytes)
#define WPACK_BYTES (KDIM * NPAD * 2)          // 720896, 256-aligned
#define APACK_OFF   WPACK_BYTES
#define APACK_ELEMS (BATCH * KDIM)             // per buffer (bf16 elems)
#define BAR_OFF     (APACK_OFF + 2 * APACK_ELEMS * 2)

__device__ __forceinline__ uint16_t f2bf(float f) {
    union { float f; unsigned u; } x; x.f = f;
    unsigned r = x.u + 0x7FFFu + ((x.u >> 16) & 1u);   // round-to-nearest-even
    return (uint16_t)(r >> 16);
}

// ---------------------------------------------------------------------------
// Setup kernel 1: pack [param2 ; param1] (logical 512 x 704, bf16) into the
// per-lane WMMA B-operand layout. Block (kc, ct) = 32x16 tile, 1024 B; lane l
// holds K = kc*32 + (l>>4)*16 + j  (j=0..15), column n = ct*16 + (l&15),
// stored contiguously so the GEMM loads two b128 per block per lane.
// Column repack: n<352 -> orig col n (inp half); n>=352 -> orig col 340+(n-352).
// ---------------------------------------------------------------------------
__global__ void setup_w_kernel(const float* __restrict__ p1,  // IN x 2H
                               const float* __restrict__ p2,  // H x 2H
                               uint16_t* __restrict__ wpack) {
    int idx = blockIdx.x * blockDim.x + threadIdx.x;
    if (idx >= KDIM * NPAD) return;
    int blk    = idx >> 9;        // /512 elems per 32x16 block
    int within = idx & 511;
    int lane   = within >> 4;
    int j      = within & 15;
    int kc = blk / NTILES_ALL;
    int ct = blk % NTILES_ALL;
    int k  = kc * 32 + ((lane >> 4) << 4) + j;   // logical K row, 0..511
    int n  = ct * 16 + (lane & 15);              // packed column, 0..703
    int half = (n >= 352) ? 1 : 0;
    int j0   = n - half * 352;
    float v = 0.0f;
    if (j0 < HID) {
        int col = half ? (HID + j0) : j0;        // column in original 2H space
        if (k < HID)
            v = p2[k * (2 * HID) + col];
        else if (k >= K_INPUT_OFF && k < K_INPUT_OFF + IN_DIM)
            v = p1[(k - K_INPUT_OFF) * (2 * HID) + col];
    }
    wpack[idx] = f2bf(v);
}

// ---------------------------------------------------------------------------
// Setup kernel 2: zero both activation buffers (state starts at 0; zeros also
// provide the K padding), place input slice t=0 into buffer 0, reset barrier.
// ---------------------------------------------------------------------------
__global__ void setup_a_kernel(const float* __restrict__ inp,
                               uint16_t* __restrict__ apack,
                               unsigned* __restrict__ bar) {
    int idx = blockIdx.x * blockDim.x + threadIdx.x;
    if (idx == 0) { bar[0] = 0u; bar[1] = 0u; }
    if (idx >= 2 * APACK_ELEMS) return;
    int buf = idx >> 16;          // / 65536
    int rem = idx & 65535;
    int b = rem >> 9;             // / 512
    int k = rem & 511;
    uint16_t v = 0;
    if (buf == 0 && k >= K_INPUT_OFF && k < K_INPUT_OFF + IN_DIM)
        v = f2bf(inp[b * IN_DIM + (k - K_INPUT_OFF)]);
    apack[idx] = v;
}

// ---------------------------------------------------------------------------
// Grid-wide sense barrier (generation counter in workspace, agent scope).
// ---------------------------------------------------------------------------
__device__ __forceinline__ void grid_barrier(unsigned* count, unsigned* gen) {
    __syncthreads();
    if (threadIdx.x == 0) {
        __threadfence();
        unsigned my = __hip_atomic_load(gen, __ATOMIC_RELAXED, __HIP_MEMORY_SCOPE_AGENT);
        unsigned prev = __hip_atomic_fetch_add(count, 1u, __ATOMIC_ACQ_REL, __HIP_MEMORY_SCOPE_AGENT);
        if (prev == NWG - 1u) {
            __hip_atomic_store(count, 0u, __ATOMIC_RELAXED, __HIP_MEMORY_SCOPE_AGENT);
            __hip_atomic_store(gen, my + 1u, __ATOMIC_RELEASE, __HIP_MEMORY_SCOPE_AGENT);
        } else {
            while (__hip_atomic_load(gen, __ATOMIC_ACQUIRE, __HIP_MEMORY_SCOPE_AGENT) == my)
                __builtin_amdgcn_s_sleep(1);
        }
    }
    __syncthreads();
}

// ---------------------------------------------------------------------------
// Persistent scan kernel. 22 WGs x 8 wave32. Workgroup g owns column-tile
// pair (c=g of inp half, c+22 of gate half); wave w owns M-tile w.
//
// Preamble: the WG's 32 weight blocks (16 k-chunks x {inp,gate}) = 32 KB are
// staged into LDS ONCE (weights are step-invariant) and read via ds_load_b128
// for all 1024 steps — eliminates the 8x-redundant per-step L2 weight reads.
//
// Per step: Y = A(t) @ W via v_wmma_f32_16x16x32_bf16 (16 K-steps x 2 halves),
// fused new_state = 1 + yi*sigmoid(yg) in registers, write fp32 out + bf16
// next-state into the other activation buffer, cooperatively repack input
// slice t+1, one grid barrier.
// ---------------------------------------------------------------------------
__global__ void __launch_bounds__(256) rnn_scan_kernel(
        const float* __restrict__ inp, const uint16_t* __restrict__ wpack,
        uint16_t* __restrict__ apack, float* __restrict__ out,
        unsigned* __restrict__ bar) {
    // LDS: 32 blocks of 1024 B: block (kc, half) at (kc*2 + half)*1024.
    __shared__ u32x4 smem[2048];                 // 32 KB

    const int c  = blockIdx.x;            // 0..21 column-tile pair
    const int m  = threadIdx.x >> 5;      // M-tile 0..7
    const int l  = threadIdx.x & 31;
    const int lhalf = l >> 4;
    const int ln    = l & 15;
    const int wid   = c * WAVES_PER_WG + m;      // 0..175

    // ---- one-time weight staging: 2048 u32x4, 8 per thread ----
    {
        const u32x4* wsrc = (const u32x4*)wpack;
#pragma unroll
        for (int j = threadIdx.x; j < 2048; j += 256) {
            int blk    = j >> 6;                 // 0..31
            int within = j & 63;
            int kc   = blk >> 1;
            int half = blk & 1;
            // src block (kc*44 + c + half*22), 64 u32x4 each
            smem[j] = wsrc[(size_t)(kc * NTILES_ALL + c + half * NTILES_HALF) * 64 + within];
        }
    }
    __syncthreads();

    // Per-lane constant byte offsets.
    const int arow_bytes = (m * 16 + ln) * (KDIM * 2);       // A row for this lane
    const int a_sub      = lhalf * 32;                       // which 16-elem chunk
    const char* lds_lane = (const char*)smem + l * 32;       // 32 B per lane per block

    unsigned* bcount = bar;
    unsigned* bgen   = bar + 1;

    for (int t = 0; t < T_STEPS; ++t) {
        const uint16_t* A     = apack + (t & 1) * APACK_ELEMS;
        uint16_t*       Anext = apack + ((t + 1) & 1) * APACK_ELEMS;

        v8f acc_i = {0.f, 0.f, 0.f, 0.f, 0.f, 0.f, 0.f, 0.f};
        v8f acc_g = {0.f, 0.f, 0.f, 0.f, 0.f, 0.f, 0.f, 0.f};

        const char* abase = (const char*)A + arow_bytes + a_sub;

#pragma unroll 4
        for (int kc = 0; kc < KTILES; ++kc) {
            union { v16bf v; u32x4 q[2]; } a, bi, bg;
            const char* ap = abase + kc * 64;
            a.q[0] = *(const u32x4*)(ap);            // K = kc*32 + lhalf*16 .. +7
            a.q[1] = *(const u32x4*)(ap + 16);       // K = kc*32 + lhalf*16 + 8 ..
            const char* lb = lds_lane + (size_t)(kc * 2) * 1024;
            bi.q[0] = *(const u32x4*)(lb);
            bi.q[1] = *(const u32x4*)(lb + 16);
            bg.q[0] = *(const u32x4*)(lb + 1024);
            bg.q[1] = *(const u32x4*)(lb + 1040);
            acc_i = __builtin_amdgcn_wmma_f32_16x16x32_bf16(
                        false, a.v, false, bi.v, (short)0, acc_i, false, false);
            acc_g = __builtin_amdgcn_wmma_f32_16x16x32_bf16(
                        false, a.v, false, bg.v, (short)0, acc_g, false, false);
        }

        // Fused elementwise: D layout => lane l, reg r -> (M = r + 8*lhalf, N = ln)
        const int h = c * 16 + ln;
        const bool hvalid = (h < HID);
#pragma unroll
        for (int r = 0; r < 8; ++r) {
            float yi = acc_i[r];
            float yg = acc_g[r];
            float s  = 1.0f + yi * (1.0f / (1.0f + __expf(-yg)));
            int b = m * 16 + r + 8 * lhalf;
            if (hvalid) {
                out[((size_t)t * BATCH + b) * HID + h] = s;
                Anext[b * KDIM + h] = f2bf(s);
            }
        }

        // Cooperatively pack input slice t+1 (all 176 waves share 128*140 elems).
        if (t + 1 < T_STEPS) {
            const float* src = inp + (size_t)(t + 1) * BATCH * IN_DIM;
            for (int idx = wid * 32 + l; idx < BATCH * IN_DIM; idx += NWG * WAVES_PER_WG * 32) {
                int b  = idx / IN_DIM;
                int cc = idx - b * IN_DIM;
                Anext[b * KDIM + K_INPUT_OFF + cc] = f2bf(src[b * IN_DIM + cc]);
            }
        }

        grid_barrier(bcount, bgen);
    }
}

extern "C" void kernel_launch(void* const* d_in, const int* in_sizes, int n_in,
                              void* d_out, int out_size, void* d_ws, size_t ws_size,
                              hipStream_t stream) {
    (void)in_sizes; (void)n_in; (void)out_size; (void)ws_size;
    const float* inp = (const float*)d_in[0];   // T x B x IN  fp32
    const float* p1  = (const float*)d_in[1];   // IN x 2H     fp32
    const float* p2  = (const float*)d_in[2];   // H  x 2H     fp32
    float* out = (float*)d_out;                 // T x B x H   fp32

    char* ws = (char*)d_ws;
    uint16_t* wpack = (uint16_t*)(ws);
    uint16_t* apack = (uint16_t*)(ws + APACK_OFF);
    unsigned* bar   = (unsigned*)(ws + BAR_OFF);

    setup_w_kernel<<<(KDIM * NPAD + 255) / 256, 256, 0, stream>>>(p1, p2, wpack);
    setup_a_kernel<<<(2 * APACK_ELEMS + 255) / 256, 256, 0, stream>>>(inp, apack, bar);
    rnn_scan_kernel<<<NWG, 256, 0, stream>>>(inp, wpack, apack, out, bar);
}
